// InteractiveGNN_55774445306359
// MI455X (gfx1250) — matrix operand order repaired and checked by reading, not compiled, not run on hardware
//
#include <hip/hip_runtime.h>

typedef __attribute__((ext_vector_type(16))) _Float16 v16h;
typedef __attribute__((ext_vector_type(8)))  float    v8f;

union AFrag { v16h h; unsigned int u[8]; };

__device__ __forceinline__ unsigned short f2h_bits(float x) {
  union { _Float16 f; unsigned short s; } u;
  u.f = (_Float16)x;
  return u.s;
}

__device__ __forceinline__ unsigned int pack2(float a, float b) {
  return (unsigned int)f2h_bits(a) | ((unsigned int)f2h_bits(b) << 16);
}

__device__ __forceinline__ v8f zero8() {
  v8f z = {0.f, 0.f, 0.f, 0.f, 0.f, 0.f, 0.f, 0.f};
  return z;
}

// ---------------------------------------------------------------------------
// EdgeConv: per wave, a 16-edge tile.
//   m = relu([h_dst | h_src | ea] @ W1 + b1) @ W2 + b2      (f16 WMMA, f32 acc)
//   agg[dst] = max(agg[dst], m)  via int-bit atomicMax on positives
//   (agg pre-zeroed => result == relu(segment_max), empty nodes -> 0)
// DH = node feature dim (16 or 64), KCHUNKS = ceil((2*DH+6)/32)
// ---------------------------------------------------------------------------
template<int DH, int KCHUNKS>
__launch_bounds__(256)
__global__ void edgeconv_kernel(const float* __restrict__ hin,
                                const float* __restrict__ ea,
                                const int*   __restrict__ srcIdx,
                                const int*   __restrict__ dstIdx,
                                const float* __restrict__ w1,
                                const float* __restrict__ b1,
                                const float* __restrict__ w2,
                                const float* __restrict__ b2,
                                float* __restrict__ agg,
                                int numTiles)
{
  constexpr int DIN  = 2 * DH + 6;
  constexpr int KPAD = KCHUNKS * 32;

  __shared__ unsigned int   wp1[(KPAD / 2) * 64];   // W1 packed as (k,k+1) half-pairs
  __shared__ unsigned int   wp2[32 * 64];           // W2 packed
  __shared__ float          b1s[64];
  __shared__ float          b2s[64];
  __shared__ unsigned short stage[8][16 * 32];      // per-wave A chunk (16 edges x 32 K halves)
  __shared__ unsigned short mid[8][16 * 64];        // per-wave hidden activations (halves)

  const int tid = threadIdx.x;

  for (int i = tid; i < (KPAD / 2) * 64; i += 256) {
    const int kp = i >> 6;
    const int n  = i & 63;
    const int k0 = 2 * kp, k1 = 2 * kp + 1;
    const float f0 = (k0 < DIN) ? w1[k0 * 64 + n] : 0.0f;
    const float f1 = (k1 < DIN) ? w1[k1 * 64 + n] : 0.0f;
    wp1[i] = pack2(f0, f1);
  }
  for (int i = tid; i < 32 * 64; i += 256) {
    const int kp = i >> 6;
    const int n  = i & 63;
    wp2[i] = pack2(w2[(2 * kp) * 64 + n], w2[(2 * kp + 1) * 64 + n]);
  }
  if (tid < 64) { b1s[tid] = b1[tid]; b2s[tid] = b2[tid]; }
  __syncthreads();

  const int wave = tid >> 5;
  const int lane = tid & 31;
  const int m16  = lane & 15;
  const int hi   = lane >> 4;
  unsigned short* const st = stage[wave];
  unsigned short* const md = mid[wave];

  for (int t = blockIdx.x * 8 + wave; t < numTiles; t += gridDim.x * 8) {
    const int e0   = t * 16;
    const int dIdx = dstIdx[e0 + m16];   // lane m (<16) holds edge m's dst
    const int sIdx = srcIdx[e0 + m16];

    v8f acc[4];
    #pragma unroll
    for (int nt = 0; nt < 4; ++nt) acc[nt] = zero8();

    #pragma unroll
    for (int c = 0; c < KCHUNKS; ++c) {
      const int kk = c * 32 + lane;      // feature this lane stages
      #pragma unroll
      for (int m = 0; m < 16; ++m) {
        const int dm = __shfl(dIdx, m, 32);
        const int sm = __shfl(sIdx, m, 32);
        float v;
        if (kk < DH)          v = hin[(size_t)dm * DH + kk];
        else if (kk < 2 * DH) v = hin[(size_t)sm * DH + (kk - DH)];
        else if (kk < DIN)    v = ea[(size_t)(e0 + m) * 6 + (kk - 2 * DH)];
        else                  v = 0.0f;
        st[m * 32 + lane] = f2h_bits(v);
      }
      __builtin_amdgcn_wave_barrier();

      AFrag a;
      #pragma unroll
      for (int j = 0; j < 8; ++j) {
        const int K0 = 2 * j + 16 * hi;
        a.u[j] = *(const unsigned int*)(st + m16 * 32 + K0);
      }
      #pragma unroll
      for (int nt = 0; nt < 4; ++nt) {
        AFrag b;
        #pragma unroll
        for (int j = 0; j < 8; ++j)
          b.u[j] = wp1[(c * 16 + 8 * hi + j) * 64 + nt * 16 + m16];
        acc[nt] = __builtin_amdgcn_wmma_f32_16x16x32_f16(
            false, a.h, false, b.h, (short)0, acc[nt], false, false);
      }
      __builtin_amdgcn_wave_barrier();
    }

    // bias + relu -> hidden halves in LDS
    #pragma unroll
    for (int nt = 0; nt < 4; ++nt) {
      const int n = nt * 16 + m16;
      const float bb = b1s[n];
      #pragma unroll
      for (int r = 0; r < 8; ++r) {
        float v = fmaxf(acc[nt][r] + bb, 0.0f);
        md[(r + 8 * hi) * 64 + n] = f2h_bits(v);
      }
    }
    __builtin_amdgcn_wave_barrier();

    v8f outA[4];
    #pragma unroll
    for (int nt = 0; nt < 4; ++nt) outA[nt] = zero8();
    #pragma unroll
    for (int c = 0; c < 2; ++c) {
      AFrag a;
      #pragma unroll
      for (int j = 0; j < 8; ++j) {
        const int K0 = 2 * j + 16 * hi;
        a.u[j] = *(const unsigned int*)(md + m16 * 64 + c * 32 + K0);
      }
      #pragma unroll
      for (int nt = 0; nt < 4; ++nt) {
        AFrag b;
        #pragma unroll
        for (int j = 0; j < 8; ++j)
          b.u[j] = wp2[(c * 16 + 8 * hi + j) * 64 + nt * 16 + m16];
        outA[nt] = __builtin_amdgcn_wmma_f32_16x16x32_f16(
            false, a.h, false, b.h, (short)0, outA[nt], false, false);
      }
    }

    // scatter-max onto dst (relu folded via zero-init; int bits order-correct for v>0)
    #pragma unroll
    for (int nt = 0; nt < 4; ++nt) {
      const int n = nt * 16 + m16;
      const float bb = b2s[n];
      #pragma unroll
      for (int r = 0; r < 8; ++r) {
        const float v = outA[nt][r] + bb;
        const int M  = r + 8 * hi;
        const int dm = __shfl(dIdx, M, 32);   // executed by all lanes (shfl outside branch)
        if (v > 0.0f)
          atomicMax((int*)(agg + (size_t)dm * 64 + n), __float_as_int(v));
      }
    }
    __builtin_amdgcn_wave_barrier();
  }
}

// ---------------------------------------------------------------------------
// Deterministic global_mean_pool: one wave per graph, binary search on sorted
// batch ids, sequential sum (no float atomics).
// ---------------------------------------------------------------------------
__launch_bounds__(256)
__global__ void pool_mean_kernel(const float* __restrict__ h,
                                 const int*   __restrict__ batch,
                                 float*       __restrict__ pooled,  // [G,64]
                                 int N, int G)
{
  const int wave = (blockIdx.x * 256 + threadIdx.x) >> 5;
  const int lane = threadIdx.x & 31;
  if (wave >= G) return;
  const int g = wave;

  int lo = 0, hb = N;
  while (lo < hb) { int m = (lo + hb) >> 1; if (batch[m] < g) lo = m + 1; else hb = m; }
  const int s = lo;
  hb = N;
  while (lo < hb) { int m = (lo + hb) >> 1; if (batch[m] < g + 1) lo = m + 1; else hb = m; }
  const int e = lo;

  float a0 = 0.0f, a1 = 0.0f;
  for (int n = s; n < e; ++n) {
    a0 += h[(size_t)n * 64 + lane];
    a1 += h[(size_t)n * 64 + 32 + lane];
  }
  const float c = fmaxf((float)(e - s), 1.0f);
  pooled[(size_t)g * 64 + lane]      = a0 / c;
  pooled[(size_t)g * 64 + 32 + lane] = a1 / c;
}

// ---------------------------------------------------------------------------
// Readout: z = relu(pooled @ Wr1 + br1) @ Wr2 + br2, WMMA, 16 graphs per wave.
// ---------------------------------------------------------------------------
__launch_bounds__(256)
__global__ void readout_kernel(const float* __restrict__ pooled,  // [G,64]
                               const float* __restrict__ wr1,     // [64,64]
                               const float* __restrict__ br1,
                               const float* __restrict__ wr2,     // [64,128]
                               const float* __restrict__ br2,
                               float* __restrict__ out,           // [G,128]
                               int numTiles)
{
  __shared__ unsigned int   wpa[32 * 64];
  __shared__ unsigned int   wpb[32 * 128];
  __shared__ float          b1s[64];
  __shared__ float          b2s[128];
  __shared__ unsigned short mid1[8][16 * 64];
  __shared__ unsigned short mid2[8][16 * 64];

  const int tid = threadIdx.x;
  for (int i = tid; i < 32 * 64; i += 256) {
    const int kp = i >> 6, n = i & 63;
    wpa[i] = pack2(wr1[(2 * kp) * 64 + n], wr1[(2 * kp + 1) * 64 + n]);
  }
  for (int i = tid; i < 32 * 128; i += 256) {
    const int kp = i >> 7, n = i & 127;
    wpb[i] = pack2(wr2[(2 * kp) * 128 + n], wr2[(2 * kp + 1) * 128 + n]);
  }
  if (tid < 64)  b1s[tid] = br1[tid];
  if (tid < 128) b2s[tid] = br2[tid];
  __syncthreads();

  const int wave = tid >> 5;
  const int lane = tid & 31;
  const int m16  = lane & 15;
  const int hi   = lane >> 4;
  unsigned short* const s1 = mid1[wave];
  unsigned short* const s2 = mid2[wave];

  const int t = blockIdx.x * 8 + wave;     // uniform per wave -> EXEC all-1 inside
  if (t < numTiles) {
    const int g0 = t * 16;
    #pragma unroll
    for (int m = 0; m < 16; ++m) {
      for (int f = lane; f < 64; f += 32)
        s1[m * 64 + f] = f2h_bits(pooled[(size_t)(g0 + m) * 64 + f]);
    }
    __builtin_amdgcn_wave_barrier();

    v8f acc[4];
    #pragma unroll
    for (int nt = 0; nt < 4; ++nt) acc[nt] = zero8();
    #pragma unroll
    for (int c = 0; c < 2; ++c) {
      AFrag a;
      #pragma unroll
      for (int j = 0; j < 8; ++j) {
        const int K0 = 2 * j + 16 * hi;
        a.u[j] = *(const unsigned int*)(s1 + m16 * 64 + c * 32 + K0);
      }
      #pragma unroll
      for (int nt = 0; nt < 4; ++nt) {
        AFrag b;
        #pragma unroll
        for (int j = 0; j < 8; ++j)
          b.u[j] = wpa[(c * 16 + 8 * hi + j) * 64 + nt * 16 + m16];
        acc[nt] = __builtin_amdgcn_wmma_f32_16x16x32_f16(
            false, a.h, false, b.h, (short)0, acc[nt], false, false);
      }
    }
    #pragma unroll
    for (int nt = 0; nt < 4; ++nt) {
      const int n = nt * 16 + m16;
      const float bb = b1s[n];
      #pragma unroll
      for (int r = 0; r < 8; ++r)
        s2[(r + 8 * hi) * 64 + n] = f2h_bits(fmaxf(acc[nt][r] + bb, 0.0f));
    }
    __builtin_amdgcn_wave_barrier();

    v8f o[8];
    #pragma unroll
    for (int nt = 0; nt < 8; ++nt) o[nt] = zero8();
    #pragma unroll
    for (int c = 0; c < 2; ++c) {
      AFrag a;
      #pragma unroll
      for (int j = 0; j < 8; ++j) {
        const int K0 = 2 * j + 16 * hi;
        a.u[j] = *(const unsigned int*)(s2 + m16 * 64 + c * 32 + K0);
      }
      #pragma unroll
      for (int nt = 0; nt < 8; ++nt) {
        AFrag b;
        #pragma unroll
        for (int j = 0; j < 8; ++j)
          b.u[j] = wpb[(c * 16 + 8 * hi + j) * 128 + nt * 16 + m16];
        o[nt] = __builtin_amdgcn_wmma_f32_16x16x32_f16(
            false, a.h, false, b.h, (short)0, o[nt], false, false);
      }
    }
    #pragma unroll
    for (int nt = 0; nt < 8; ++nt) {
      const int n = nt * 16 + m16;
      const float bb = b2s[n];
      #pragma unroll
      for (int r = 0; r < 8; ++r)
        out[(size_t)(g0 + r + 8 * hi) * 128 + n] = o[nt][r] + bb;
    }
  }
}

// ---------------------------------------------------------------------------
extern "C" void kernel_launch(void* const* d_in, const int* in_sizes, int n_in,
                              void* d_out, int out_size, void* d_ws, size_t ws_size,
                              hipStream_t stream) {
  const float* x     = (const float*)d_in[0];
  const float* ea    = (const float*)d_in[1];
  const int*   eidx  = (const int*)  d_in[2];
  const int*   batch = (const int*)  d_in[3];
  const float* w11   = (const float*)d_in[4];
  const float* b11   = (const float*)d_in[5];
  const float* w12   = (const float*)d_in[6];
  const float* b12   = (const float*)d_in[7];
  const float* w21   = (const float*)d_in[8];
  const float* b21   = (const float*)d_in[9];
  const float* w22   = (const float*)d_in[10];
  const float* b22   = (const float*)d_in[11];
  const float* wr1   = (const float*)d_in[12];
  const float* br1   = (const float*)d_in[13];
  const float* wr2   = (const float*)d_in[14];
  const float* br2   = (const float*)d_in[15];

  const int N = in_sizes[0] / 16;   // 131072
  const int E = in_sizes[2] / 2;    // 2097152
  const int G = 1024;

  const int* srcIdx = eidx;         // edge_index[0]
  const int* dstIdx = eidx + E;     // edge_index[1]

  float* agg1   = (float*)d_ws;                 // [N,64]
  float* agg2   = agg1 + (size_t)N * 64;        // [N,64]
  float* pooled = agg2 + (size_t)N * 64;        // [G,64]

  // Zero both scatter-max accumulators (ReLU identity) every call -> replay-safe.
  hipMemsetAsync(d_ws, 0, (size_t)N * 64 * 2 * sizeof(float), stream);

  const int numTiles = E / 16;
  edgeconv_kernel<16, 2><<<dim3(1024), dim3(256), 0, stream>>>(
      x, ea, srcIdx, dstIdx, w11, b11, w12, b12, agg1, numTiles);
  edgeconv_kernel<64, 5><<<dim3(1024), dim3(256), 0, stream>>>(
      agg1, ea, srcIdx, dstIdx, w21, b21, w22, b22, agg2, numTiles);

  pool_mean_kernel<<<dim3((G + 7) / 8), dim3(256), 0, stream>>>(agg2, batch, pooled, N, G);

  readout_kernel<<<dim3((G / 16 + 7) / 8), dim3(256), 0, stream>>>(
      pooled, wr1, br1, wr2, br2, (float*)d_out, G / 16);
}
